// GATModel_3126736192225
// MI455X (gfx1250) — compile-verified, matrix operand order checked by reading, not run on hardware
//
#include <hip/hip_runtime.h>
#include <hip/hip_bf16.h>
#include <float.h>

// ---------------------------------------------------------------------------
// GAT (3 layers, H=4, C=64) + MLP head for MI455X (gfx1250, wave32, WMMA).
// Dense GEMMs: v_wmma_f32_16x16x32_bf16, weights staged in LDS via
// global_load_async_to_lds_b128 (ASYNCcnt), B-fragments via ds_load_b128.
// Edge softmax/aggregation uses L2-resident f32 atomics with bf16 gathers.
// ---------------------------------------------------------------------------

#define GAT_N   50000
#define GAT_E   800000
#define GAT_ET  (GAT_E + GAT_N)   // edges + self loops
#define GAT_H   4
#define GAT_C   64
#define GAT_HC  256
#define GAT_IN  61
#define NEG_SLOPE 0.2f

typedef __bf16 bf16_t;
typedef __attribute__((ext_vector_type(16))) __bf16 v16bf;
typedef __attribute__((ext_vector_type(8)))  float  v8f;

// ------------------------- small helpers -----------------------------------

__device__ inline unsigned ford(float f) {           // order-preserving encode
    unsigned u = __float_as_uint(f);
    return (u & 0x80000000u) ? ~u : (u | 0x80000000u);
}
__device__ inline float fdec(unsigned u) {
    return __uint_as_float((u & 0x80000000u) ? (u ^ 0x80000000u) : ~u);
}
__device__ inline float lrelu(float v) { return v > 0.f ? v : NEG_SLOPE * v; }

__device__ inline void edge_sd(const long long* __restrict__ ei, int e,
                               int& s, int& d) {
    if (e < GAT_E) { s = (int)ei[e]; d = (int)ei[GAT_E + e]; }
    else           { s = d = e - GAT_E; }
}

// Load one 16-bit 16x32 WMMA fragment slice for this lane.
// Per ISA 7.12.2: half-lane h holds elements [k0+8h, k0+8h+8) and
// [k0+16+8h, k0+16+8h+8) -> two aligned 16B chunks (b128 loads).
__device__ inline v16bf load_frag(const bf16_t* __restrict__ base, int k0, int half) {
    union { v16bf v; uint4 q[2]; } f;
    f.q[0] = *(const uint4*)(base + k0 + 8 * half);
    f.q[1] = *(const uint4*)(base + k0 + 16 + 8 * half);
    return f.v;
}

// ------------------------- prep / fill kernels ------------------------------

__global__ void fill_u32_kernel(unsigned* __restrict__ p, unsigned val, long n) {
    long i = (long)blockIdx.x * blockDim.x + threadIdx.x;
    if (i < n) p[i] = val;
}

// x fp32 [N,61] -> bf16 [N,64], zero padded
__global__ void prep_x_kernel(const float* __restrict__ x, bf16_t* __restrict__ xb) {
    long i = (long)blockIdx.x * blockDim.x + threadIdx.x;
    if (i >= (long)GAT_N * 64) return;
    int r = (int)(i >> 6), k = (int)(i & 63);
    xb[i] = (k < GAT_IN) ? (bf16_t)x[(long)r * GAT_IN + k] : (bf16_t)0.0f;
}

// W fp32 [din][dout] -> Wt bf16 [dout][kpad] (transposed, K zero-padded)
__global__ void transpose_w_kernel(const float* __restrict__ W, bf16_t* __restrict__ Wt,
                                   int din, int dout, int kpad) {
    long i = (long)blockIdx.x * blockDim.x + threadIdx.x;
    if (i >= (long)dout * kpad) return;
    int n = (int)(i / kpad), k = (int)(i % kpad);
    Wt[i] = (k < din) ? (bf16_t)W[(long)k * dout + n] : (bf16_t)0.0f;
}

// ------------------------- WMMA GEMM (LDS-staged B) -------------------------
// out[rows][dout] = A[rows][kdim](bf16) @ Wt[dout][kdim](bf16)^T, f32 accum.
// Block = 256 thr = 8 waves; each wave: 16 rows x 64 cols (4 wmma tiles).
// The 64-col weight tile (<=32KB) is copied once per block into LDS with
// GLOBAL_LOAD_ASYNC_TO_LDS_B128 (ASYNCcnt) and re-read via ds_load_b128.

__global__ __launch_bounds__(256) void gemm_bf16_kernel(
    const bf16_t* __restrict__ A, int lda,
    const bf16_t* __restrict__ Wt,           // [dout][kdim]
    float* __restrict__ outF,                // optional
    bf16_t* __restrict__ outB,               // optional
    int rows, int kdim, int dout)
{
    __shared__ bf16_t lb[64 * 256];          // weight tile [64 cols][kdim]

    const int tid   = threadIdx.x;
    const int wave  = tid >> 5;
    const int lane  = tid & 31;
    const int l16   = lane & 15;
    const int half  = lane >> 4;
    const int baseRow = blockIdx.x * 128 + wave * 16;
    const int col0    = blockIdx.y * 64;

    // ---- async copy: 64*kdim bf16 (contiguous rows of Wt) -> LDS ----
    {
        const unsigned tileBytes = (unsigned)(64 * kdim * 2);
        const char* gsrc = (const char*)(Wt + (long)col0 * kdim);
        unsigned ldsBase = (unsigned)(size_t)(__attribute__((address_space(3))) char*)lb;
        for (unsigned off = (unsigned)tid * 16u; off < tileBytes; off += 256u * 16u) {
            unsigned dstLds = ldsBase + off;
            const char* src = gsrc + off;
            asm volatile("global_load_async_to_lds_b128 %0, %1, off"
                         :: "v"(dstLds), "v"(src) : "memory");
        }
        asm volatile("s_wait_asynccnt 0" ::: "memory");
        __syncthreads();
    }

    int mA = baseRow + l16; if (mA >= rows) mA = rows - 1;
    const bf16_t* Arow = A + (long)mA * lda;

    v8f acc[4];
#pragma unroll
    for (int t = 0; t < 4; ++t) acc[t] = (v8f){0.f,0.f,0.f,0.f,0.f,0.f,0.f,0.f};

    for (int k0 = 0; k0 < kdim; k0 += 32) {
        __builtin_prefetch(Arow + k0 + 32, 0, 3);   // WGP-scope prefetch, A stream
        v16bf afv = load_frag(Arow, k0, half);      // global_load_b128 x2
        v16bf bfv[4];
#pragma unroll
        for (int t = 0; t < 4; ++t)                 // ds_load_b128 x2 each
            bfv[t] = load_frag(lb + (t * 16 + l16) * kdim, k0, half);
#pragma unroll
        for (int t = 0; t < 4; ++t)
            acc[t] = __builtin_amdgcn_wmma_f32_16x16x32_bf16(
                false, afv, false, bfv[t], (short)0, acc[t], false, false);
    }

#pragma unroll
    for (int t = 0; t < 4; ++t) {
        int col = col0 + t * 16 + l16;
#pragma unroll
        for (int r = 0; r < 8; ++r) {
            int row = baseRow + r + 8 * half;
            if (row < rows) {
                float v = acc[t][r];
                if (outF) outF[(long)row * dout + col] = v;
                if (outB) outB[(long)row * dout + col] = (bf16_t)v;
            }
        }
    }
}

// ------------------------- attention logits ---------------------------------
// One wave per node: lane owns channels [8L, 8L+8) -> head = L/8.
__global__ void attn_logits_kernel(const bf16_t* __restrict__ h,
                                   const float* __restrict__ att_src,
                                   const float* __restrict__ att_dst,
                                   float* __restrict__ a_src,
                                   float* __restrict__ a_dst)
{
    int node = (int)(((long)blockIdx.x * blockDim.x + threadIdx.x) >> 5);
    int lane = threadIdx.x & 31;
    if (node >= GAT_N) return;
    int head  = lane >> 3;
    int cbase = lane * 8;
    const bf16_t* hp = h + (long)node * GAT_HC + cbase;
    float ps = 0.f, pd = 0.f;
#pragma unroll
    for (int j = 0; j < 8; ++j) {
        float v = (float)hp[j];
        ps += v * att_src[cbase + j];
        pd += v * att_dst[cbase + j];
    }
#pragma unroll
    for (int off = 1; off < 8; off <<= 1) {
        ps += __shfl_xor(ps, off, 32);
        pd += __shfl_xor(pd, off, 32);
    }
    if ((lane & 7) == 0) {
        a_src[node * GAT_H + head] = ps;
        a_dst[node * GAT_H + head] = pd;
    }
}

// ------------------------- edge softmax phases ------------------------------

__global__ void edge_max_kernel(const long long* __restrict__ ei,
                                const float* __restrict__ a_src,
                                const float* __restrict__ a_dst,
                                unsigned* __restrict__ mU)
{
    int e = (int)((long)blockIdx.x * blockDim.x + threadIdx.x);
    if (e >= GAT_ET) return;
    int s, d; edge_sd(ei, e, s, d);
#pragma unroll
    for (int h = 0; h < GAT_H; ++h) {
        float v = lrelu(a_src[s * GAT_H + h] + a_dst[d * GAT_H + h]);
        atomicMax(&mU[d * GAT_H + h], ford(v));
    }
}

__global__ void edge_expsum_kernel(const long long* __restrict__ ei,
                                   const float* __restrict__ a_src,
                                   const float* __restrict__ a_dst,
                                   const unsigned* __restrict__ mU,
                                   float* __restrict__ denom)
{
    int e = (int)((long)blockIdx.x * blockDim.x + threadIdx.x);
    if (e >= GAT_ET) return;
    int s, d; edge_sd(ei, e, s, d);
#pragma unroll
    for (int h = 0; h < GAT_H; ++h) {
        float v = lrelu(a_src[s * GAT_H + h] + a_dst[d * GAT_H + h]);
        float m = fdec(mU[d * GAT_H + h]);
        atomicAdd(&denom[d * GAT_H + h], __expf(v - m));
    }
}

// One wave per edge: lane owns 8 channels of head lane/8.
__global__ void aggregate_kernel(const long long* __restrict__ ei,
                                 const float* __restrict__ a_src,
                                 const float* __restrict__ a_dst,
                                 const unsigned* __restrict__ mU,
                                 const float* __restrict__ denom,
                                 const bf16_t* __restrict__ h,
                                 float* __restrict__ out)
{
    int e = (int)(((long)blockIdx.x * blockDim.x + threadIdx.x) >> 5);
    int lane = threadIdx.x & 31;
    if (e >= GAT_ET) return;
    int s = 0, d = 0;
    if (lane == 0) edge_sd(ei, e, s, d);     // one load per wave,
    s = __shfl(s, 0, 32);                    // broadcast to all lanes
    d = __shfl(d, 0, 32);
    int head = lane >> 3;
    float v = lrelu(a_src[s * GAT_H + head] + a_dst[d * GAT_H + head]);
    float m = fdec(mU[d * GAT_H + head]);
    float alpha = __expf(v - m) / denom[d * GAT_H + head];
    int cbase = lane * 8;
    const bf16_t* hp = h + (long)s * GAT_HC + cbase;
    float* op = out + (long)d * GAT_HC + cbase;
#pragma unroll
    for (int j = 0; j < 8; ++j)
        atomicAdd(op + j, (float)hp[j] * alpha);
}

// ------------------------- bias + ELU -> next-layer bf16 input --------------

__global__ void bias_elu_kernel(const float* __restrict__ agg,
                                const float* __restrict__ bias,
                                bf16_t* __restrict__ y)
{
    long i = (long)blockIdx.x * blockDim.x + threadIdx.x;
    if (i >= (long)GAT_N * GAT_HC) return;
    float v = agg[i] + bias[i & (GAT_HC - 1)];
    v = (v > 0.f) ? v : (__expf(v) - 1.0f);   // ELU(alpha=1)
    y[i] = (bf16_t)v;
}

// ------------------------- MLP head tail ------------------------------------

__global__ void head_tail_kernel(const float* __restrict__ t,
                                 const float* __restrict__ b1,
                                 const float* __restrict__ w2,
                                 const float* __restrict__ b2,
                                 float* __restrict__ out)
{
    int n = (int)((long)blockIdx.x * blockDim.x + threadIdx.x);
    if (n >= GAT_N) return;
    float acc = b2[0];
    const float* tp = t + (long)n * 64;
#pragma unroll 8
    for (int j = 0; j < 64; ++j) {
        float v = tp[j] + b1[j];
        if (v > 0.f) acc += v * w2[j];
    }
    out[n] = acc;
}

// ---------------------------------------------------------------------------

static inline unsigned cdiv_l(long a, long b) { return (unsigned)((a + b - 1) / b); }

extern "C" void kernel_launch(void* const* d_in, const int* in_sizes, int n_in,
                              void* d_out, int out_size, void* d_ws, size_t ws_size,
                              hipStream_t stream) {
    (void)in_sizes; (void)n_in; (void)out_size; (void)ws_size;

    const float*     x  = (const float*)d_in[0];
    const long long* ei = (const long long*)d_in[1];
    const float* Ws[3]  = {(const float*)d_in[2], (const float*)d_in[6], (const float*)d_in[10]};
    const float* ASv[3] = {(const float*)d_in[3], (const float*)d_in[7], (const float*)d_in[11]};
    const float* ADv[3] = {(const float*)d_in[4], (const float*)d_in[8], (const float*)d_in[12]};
    const float* Bv[3]  = {(const float*)d_in[5], (const float*)d_in[9], (const float*)d_in[13]};
    const float* head_w1 = (const float*)d_in[14];
    const float* head_b1 = (const float*)d_in[15];
    const float* head_w2 = (const float*)d_in[16];
    const float* head_b2 = (const float*)d_in[17];
    float* out = (float*)d_out;

    // ---- workspace carve (all offsets 256B-aligned) ----
    char* p = (char*)d_ws;
    bf16_t*  bufA  = (bf16_t*)p;  p += (size_t)GAT_N * GAT_HC * 2;   // layer input (bf16)
    bf16_t*  bufB  = (bf16_t*)p;  p += (size_t)GAT_N * GAT_HC * 2;   // h = x@W (bf16)
    float*   agg   = (float*)p;   p += (size_t)GAT_N * GAT_HC * 4;   // aggregated out (f32)
    float*   a_src = (float*)p;   p += (size_t)GAT_N * GAT_H * 4;
    float*   a_dst = (float*)p;   p += (size_t)GAT_N * GAT_H * 4;
    unsigned* mU   = (unsigned*)p; p += (size_t)GAT_N * GAT_H * 4;
    float*   denom = (float*)p;   p += (size_t)GAT_N * GAT_H * 4;
    float*   tHead = (float*)p;   p += (size_t)GAT_N * 64 * 4;
    bf16_t*  wtBuf = (bf16_t*)p;  p += (size_t)GAT_HC * GAT_HC * 2;

    const int dins[3]  = {GAT_IN, GAT_HC, GAT_HC};
    const int kpads[3] = {64, GAT_HC, GAT_HC};
    const unsigned NEG_INF_ENC = 0x00800000u;  // ford(-FLT_MAX)

    // layer-0 input: x -> bf16 [N,64] padded
    prep_x_kernel<<<cdiv_l((long)GAT_N * 64, 256), 256, 0, stream>>>(x, bufA);

    for (int l = 0; l < 3; ++l) {
        const int kpad = kpads[l];

        // weights: transpose + bf16
        transpose_w_kernel<<<cdiv_l((long)GAT_HC * kpad, 256), 256, 0, stream>>>(
            Ws[l], wtBuf, dins[l], GAT_HC, kpad);

        // h = in @ W  (WMMA, LDS-staged weights)
        {
            dim3 grid(cdiv_l(GAT_N, 128), GAT_HC / 64);
            gemm_bf16_kernel<<<grid, 256, 0, stream>>>(
                bufA, kpad, wtBuf, nullptr, bufB, GAT_N, kpad, GAT_HC);
        }

        // per-node attention logits
        attn_logits_kernel<<<cdiv_l((long)GAT_N * 32, 256), 256, 0, stream>>>(
            bufB, ASv[l], ADv[l], a_src, a_dst);

        // init segment-max / denom / accumulator
        fill_u32_kernel<<<cdiv_l((long)GAT_N * GAT_H, 256), 256, 0, stream>>>(
            mU, NEG_INF_ENC, (long)GAT_N * GAT_H);
        fill_u32_kernel<<<cdiv_l((long)GAT_N * GAT_H, 256), 256, 0, stream>>>(
            (unsigned*)denom, 0u, (long)GAT_N * GAT_H);
        fill_u32_kernel<<<cdiv_l((long)GAT_N * GAT_HC, 256), 256, 0, stream>>>(
            (unsigned*)agg, 0u, (long)GAT_N * GAT_HC);

        // edge softmax: max, exp-sum, weighted scatter-aggregate
        edge_max_kernel<<<cdiv_l(GAT_ET, 256), 256, 0, stream>>>(ei, a_src, a_dst, mU);
        edge_expsum_kernel<<<cdiv_l(GAT_ET, 256), 256, 0, stream>>>(ei, a_src, a_dst, mU, denom);
        aggregate_kernel<<<cdiv_l((long)GAT_ET * 32, 256), 256, 0, stream>>>(
            ei, a_src, a_dst, mU, denom, bufB, agg);

        // bias + ELU -> next layer's bf16 input (bufA, lda 256)
        bias_elu_kernel<<<cdiv_l((long)GAT_N * GAT_HC, 256), 256, 0, stream>>>(
            agg, Bv[l], bufA);
    }

    // MLP head: t = y @ head_w1  (WMMA), then relu-dot with head_w2
    transpose_w_kernel<<<cdiv_l((long)64 * GAT_HC, 256), 256, 0, stream>>>(
        head_w1, wtBuf, GAT_HC, 64, GAT_HC);
    {
        dim3 grid(cdiv_l(GAT_N, 128), 1);
        gemm_bf16_kernel<<<grid, 256, 0, stream>>>(
            bufA, GAT_HC, wtBuf, tHead, nullptr, GAT_N, GAT_HC, 64);
    }
    head_tail_kernel<<<cdiv_l(GAT_N, 256), 256, 0, stream>>>(
        tHead, head_b1, head_w2, head_b2, out);
}
